// CfCActorWithSpike_85057532330471
// MI455X (gfx1250) — matrix operand order repaired and checked by reading, not compiled, not run on hardware
//
#include <hip/hip_runtime.h>
#include <hip/hip_bf16.h>
#include <stdint.h>

typedef __attribute__((ext_vector_type(16))) _Float16 v16h;
typedef __attribute__((ext_vector_type(8)))  float    v8f;

#define B_SZ   4096
#define T_SZ   512
#define IN_SZ  8
#define SIG_MIN 0.05f
#define SIG_MAX 2.0f
#define LOG2E  1.4426950408889634f

struct Params {
    const float*   x;
    const float*   hx;
    const float*   log_std;
    const float*   w[3][4];     // [layer][ff1, ff2, ta, tb]
    const float*   bias[3][4];
    const uint8_t* mask[3];
};

// Hardware-trans versions: v_exp_f32 (exp2) + v_rcp_f32, no IEEE div expansion.
__device__ __forceinline__ float fast_sigmoid(float x) {
    float e = __builtin_amdgcn_exp2f(-LOG2E * x);
    return __builtin_amdgcn_rcpf(1.0f + e);
}
__device__ __forceinline__ float fast_tanh(float x) {
    // tanh(x) = 1 - 2/(exp(2x)+1)
    float e = __builtin_amdgcn_exp2f((2.0f * LOG2E) * x);
    float r = __builtin_amdgcn_rcpf(e + 1.0f);
    return __builtin_fmaf(-2.0f, r, 1.0f);
}
__device__ __forceinline__ v8f wmma16(v16h a, v16h b, v8f c) {
    // D = A(16x32 f16) * B(32x16 f16) + C(16x16 f32)
    return __builtin_amdgcn_wmma_f32_16x16x32_f16(
        false, a, false, b, (short)0, c, false, false);
}

// A-layout (16-bit 16x32): lane m=l&15 is row M; lanes<16 hold K=0..15 (j=0..15)
// and lanes>=16 hold K=16..31 (j=0..15), same contiguous order as B.
// B-layout (16-bit 32x16): lane l holds column N=l&15; lanes<16 hold K=0..15
// (j=0..15), lanes>=16 hold K=16..31.
// C/D layout (f32 16x16): lane l holds N=l&15; VGPR r holds M=r+8*(l>=16).

__global__ void __launch_bounds__(256) cfc_kernel(Params p, float* __restrict__ out) {
    const int lane  = threadIdx.x & 31;
    const int wave  = blockIdx.x * (blockDim.x >> 5) + (threadIdx.x >> 5);
    const int m     = lane & 15;          // A row / batch-in-tile
    const bool lo   = lane < 16;
    const int b     = wave * 16 + m;      // this lane's batch sequence

    // ---------------- stationary weights in A layout, bias folded at K==c ---
    const int HS[3] = {9, 5, 2};
    const int CS[3] = {17, 14, 7};
    v16h aF1[3], aF2[3], aT[3];
    const int kb = lo ? 0 : 16;
    #pragma unroll
    for (int i = 0; i < 3; ++i) {
        const int h = HS[i], c = CS[i];
        v16h f1, f2, tt;
        #pragma unroll
        for (int j = 0; j < 16; ++j) {
            const int K = kb + j;
            float vf1 = 0.f, vf2 = 0.f, vt = 0.f;
            if (m < h) {
                if (K < c) {
                    const float msk = p.mask[i][m * c + K] ? 1.f : 0.f;
                    vf1 = p.w[i][0][m * c + K] * msk;
                    vf2 = p.w[i][1][m * c + K] * msk;
                    vt  = p.w[i][2][m * c + K] + p.w[i][3][m * c + K]; // ta+tb fused
                } else if (K == c) {  // folded bias column (B supplies 1.0 here)
                    vf1 = p.bias[i][0][m];
                    vf2 = p.bias[i][1][m];
                    vt  = p.bias[i][2][m] + p.bias[i][3][m];
                }
            }
            f1[j] = (_Float16)vf1; f2[j] = (_Float16)vf2; tt[j] = (_Float16)vt;
        }
        aF1[i] = f1; aF2[i] = f2; aT[i] = tt;
    }

    // ---------------- recurrent state in D layout (f32, 8 rows per lane) ----
    v8f h0 = {0,0,0,0,0,0,0,0}, h1 = {0,0,0,0,0,0,0,0}, h2 = {0,0,0,0,0,0,0,0};
    {
        const float* hxb = p.hx + (size_t)b * 16;
        if (lo) {
            #pragma unroll
            for (int j = 0; j < 8; ++j) h0[j] = hxb[j];        // h0 rows 0..7
            #pragma unroll
            for (int j = 0; j < 5; ++j) h1[j] = hxb[9 + j];    // h1 rows 0..4
            h2[0] = hxb[14]; h2[1] = hxb[15];
        } else {
            h0[0] = hxb[8];                                    // h0 row 8
        }
    }

    const float stdv = __builtin_amdgcn_exp2f(LOG2E * p.log_std[0]);
    const float* xrow = p.x + (size_t)b * T_SZ * IN_SZ;
    float* omu  = out;
    float* ostd = out + (size_t)B_SZ * T_SZ;
    float* ospk = out + 2 * (size_t)B_SZ * T_SZ;
    const _Float16 ONE  = (_Float16)1.0f;
    const _Float16 ZERO = (_Float16)0.0f;
    const v8f zc = {0,0,0,0,0,0,0,0};

    // Software pipeline: x_t held in registers entering step t; x_{t+1} issued
    // right after b0 is built so its wait lands a full step body later.
    float4 xa = {0,0,0,0}, xb = {0,0,0,0};
    if (lo) {
        const float4* xp = reinterpret_cast<const float4*>(xrow);
        xa = xp[0]; xb = xp[1];
    }

    for (int t = 0; t < T_SZ; ++t) {
        // -------- layer 0 : xx = [x(8), h0(9), 1]  (K = 18) -----------------
        v16h b0;
        if (lo) {
            b0[0] = (_Float16)xa.x; b0[1] = (_Float16)xa.y;
            b0[2] = (_Float16)xa.z; b0[3] = (_Float16)xa.w;
            b0[4] = (_Float16)xb.x; b0[5] = (_Float16)xb.y;
            b0[6] = (_Float16)xb.z; b0[7] = (_Float16)xb.w;
            #pragma unroll
            for (int j = 0; j < 8; ++j) b0[8 + j] = (_Float16)h0[j];  // K=8..15
        } else {
            b0[0] = (_Float16)h0[0];   // K=16 -> h0[8] (upper half-wave holds row 8)
            b0[1] = ONE;               // K=17 -> bias slot
            #pragma unroll
            for (int j = 2; j < 16; ++j) b0[j] = ZERO;
        }
        // issue next step's x load now; consumed only at the top of t+1
        if (lo && (t + 1 < T_SZ)) {
            const float4* xpn =
                reinterpret_cast<const float4*>(xrow + (size_t)(t + 1) * IN_SZ);
            xa = xpn[0]; xb = xpn[1];
            if (t + 8 < T_SZ)
                __builtin_prefetch(xrow + (size_t)(t + 8) * IN_SZ, 0, 1);
        }
        v8f d1 = wmma16(aF1[0], b0, zc);
        v8f d2 = wmma16(aF2[0], b0, zc);
        v8f dt = wmma16(aT [0], b0, zc);
        #pragma unroll
        for (int j = 0; j < 8; ++j) {
            float f1 = fast_tanh(d1[j]);
            float f2 = fast_tanh(d2[j]);
            float ti = fast_sigmoid(dt[j]);
            h0[j] = f1 + ti * (f2 - f1);
        }

        // -------- layer 1 : xx = [h0(9), h1(5), 1]  (K = 15) ----------------
        float h08 = __shfl_xor(h0[0], 16, 32);     // lanes<16 receive h0[8]
        v16h b1;
        if (lo) {
            #pragma unroll
            for (int j = 0; j < 8; ++j) b1[j] = (_Float16)h0[j];
            b1[8] = (_Float16)h08;
            #pragma unroll
            for (int j = 0; j < 5; ++j) b1[9 + j] = (_Float16)h1[j];
            b1[14] = ONE;                           // bias slot (K=14)
            b1[15] = ZERO;
        } else {
            #pragma unroll
            for (int j = 0; j < 16; ++j) b1[j] = ZERO;  // K>=16 all zero
        }
        d1 = wmma16(aF1[1], b1, zc);
        d2 = wmma16(aF2[1], b1, zc);
        dt = wmma16(aT [1], b1, zc);
        #pragma unroll
        for (int j = 0; j < 5; ++j) {
            float f1 = fast_tanh(d1[j]);
            float f2 = fast_tanh(d2[j]);
            float ti = fast_sigmoid(dt[j]);
            h1[j] = f1 + ti * (f2 - f1);
        }

        // -------- layer 2 : xx = [h1(5), h2(2), 1]  (K = 8) -----------------
        v16h b2;
        if (lo) {
            #pragma unroll
            for (int j = 0; j < 5; ++j) b2[j] = (_Float16)h1[j];
            b2[5] = (_Float16)h2[0];
            b2[6] = (_Float16)h2[1];
            b2[7] = ONE;                            // bias slot (K=7)
            #pragma unroll
            for (int j = 8; j < 16; ++j) b2[j] = ZERO;
        } else {
            #pragma unroll
            for (int j = 0; j < 16; ++j) b2[j] = ZERO;
        }
        d1 = wmma16(aF1[2], b2, zc);
        d2 = wmma16(aF2[2], b2, zc);
        dt = wmma16(aT [2], b2, zc);
        #pragma unroll
        for (int j = 0; j < 2; ++j) {
            float f1 = fast_tanh(d1[j]);
            float f2 = fast_tanh(d2[j]);
            float ti = fast_sigmoid(dt[j]);
            h2[j] = f1 + ti * (f2 - f1);
        }

        // -------- per-step outputs ------------------------------------------
        if (lo) {
            const size_t o = (size_t)b * T_SZ + t;
            omu [o] = fast_sigmoid(h2[0]) * (SIG_MAX - SIG_MIN) + SIG_MIN;
            ostd[o] = stdv;
            ospk[o] = h2[1];
        }
    }

    // ---------------- hx_new = concat(h0, h1, h2) ---------------------------
    float* ohx = out + 3 * (size_t)B_SZ * T_SZ + (size_t)b * 16;
    if (lo) {
        #pragma unroll
        for (int j = 0; j < 8; ++j) ohx[j] = h0[j];
        #pragma unroll
        for (int j = 0; j < 5; ++j) ohx[9 + j] = h1[j];
        ohx[14] = h2[0]; ohx[15] = h2[1];
    } else {
        ohx[8] = h0[0];
    }
}

extern "C" void kernel_launch(void* const* d_in, const int* in_sizes, int n_in,
                              void* d_out, int out_size, void* d_ws, size_t ws_size,
                              hipStream_t stream) {
    (void)in_sizes; (void)n_in; (void)out_size; (void)d_ws; (void)ws_size;
    Params p;
    p.x       = (const float*)d_in[0];
    p.hx      = (const float*)d_in[1];
    p.log_std = (const float*)d_in[2];
    for (int i = 0; i < 3; ++i) {
        const int base = 3 + 9 * i;   // [ff1_w, ff1_b, ff2_w, ff2_b, ta_w, ta_b, tb_w, tb_b, mask]
        for (int k = 0; k < 4; ++k) {
            p.w[i][k]    = (const float*)d_in[base + 2 * k];
            p.bias[i][k] = (const float*)d_in[base + 2 * k + 1];
        }
        p.mask[i] = (const uint8_t*)d_in[base + 8];
    }
    // 4096 sequences / 16 per wave = 256 waves; 8 waves (256 threads) per block.
    dim3 block(256);
    dim3 grid((B_SZ / 16) / 8);
    hipLaunchKernelGGL(cfc_kernel, grid, block, 0, stream, p, (float*)d_out);
}
